// MultiheadAttention_78632261255702
// MI455X (gfx1250) — matrix operand order repaired and checked by reading, not compiled, not run on hardware
//
#include <hip/hip_runtime.h>
#include <hip/hip_bf16.h>
#include <stdint.h>

typedef __attribute__((ext_vector_type(16))) __bf16 v16bf;
typedef __attribute__((ext_vector_type(8)))  float  v8f;

#define L_DIM   1024
#define S_DIM   1024
#define N_B     16
#define E_DIM   256
#define H_HEADS 8
#define D_HEAD  32
#define NH      128
#define SCALING 0.1767766952966369f   // 32^-0.5
#define BC      64                    // S tile
#define WAVES   8

// ---- ISA 7.12.2 layout helpers (wave32) -------------------------------
// A (16x32 bf16): lane%16 = M row; element e -> K = e + half*16 (contiguous order chosen here, not ISA's)
// B (32x16 bf16): lane%16 = N col; element e -> K = e + half*16 (same contiguous order as A, chosen here)
// C/D (16x16 f32): element r -> M = r + half*8, N = lane%16
__device__ __forceinline__ int a_k(int e, int hlf) { return e + hlf * 16; }

__device__ __forceinline__ v8f wmma_bf16(v16bf a, v16bf b, v8f c) {
    return __builtin_amdgcn_wmma_f32_16x16x32_bf16(false, a, false, b, (short)0, c, false, false);
}

// ======================================================================
// Kernel 1: fused QKV projection.  out[r,j] = X[r,:]·W[j,:] + b[j] (q scaled)
// Q,K written head-major [b=n*H+h][pos][d]; V written TRANSPOSED
// head-major [b][d][pos] so kernel 2's V B-operand reads are contiguous.
// ======================================================================
__global__ __launch_bounds__(256) void qkv_proj_kernel(
    const float* __restrict__ query, const float* __restrict__ keyi,
    const float* __restrict__ vali,  const float* __restrict__ w,
    const float* __restrict__ bias,
    __bf16* __restrict__ qh, __bf16* __restrict__ kh, __bf16* __restrict__ vh)
{
    const int which = blockIdx.y;                       // 0=q 1=k 2=v
    const float* X = (which == 0) ? query : (which == 1 ? keyi : vali);
    const float* W = w + (size_t)which * E_DIM * E_DIM;
    const float* B = bias + which * E_DIM;
    __bf16* OUT = (which == 0) ? qh : (which == 1 ? kh : vh);
    const float scale = (which == 0) ? SCALING : 1.0f;

    const int wave = threadIdx.x >> 5;
    const int lane = threadIdx.x & 31;
    const int hlf  = lane >> 4;
    const int l16  = lane & 15;
    const int r0   = blockIdx.x * 128 + wave * 16;      // 16 token-rows per wave

    for (int n0 = 0; n0 < E_DIM; n0 += 64) {
        v8f acc[4] = {};
        for (int k0 = 0; k0 < E_DIM; k0 += 32) {
            v16bf a;
            #pragma unroll
            for (int e = 0; e < 16; ++e)
                a[e] = (__bf16)X[(r0 + l16) * E_DIM + k0 + a_k(e, hlf)];
            #pragma unroll
            for (int nt = 0; nt < 4; ++nt) {
                const int j = n0 + nt * 16 + l16;
                v16bf bm;
                #pragma unroll
                for (int e = 0; e < 16; ++e)
                    bm[e] = (__bf16)W[j * E_DIM + k0 + e + hlf * 16];
                acc[nt] = wmma_bf16(a, bm, acc[nt]);
            }
        }
        #pragma unroll
        for (int nt = 0; nt < 4; ++nt) {
            #pragma unroll
            for (int r = 0; r < 8; ++r) {
                const int row = r0 + r + hlf * 8;       // token row = pos*N + n
                const int col = n0 + nt * 16 + l16;     // embed col = h*D + d
                const float v = (acc[nt][r] + B[col]) * scale;
                const int p = row >> 4, n = row & 15;   // pos, batch
                const int h = col >> 5, d = col & 31;
                const size_t b = (size_t)(n * H_HEADS + h);
                if (which == 2)
                    OUT[(b * D_HEAD + d) * 1024 + p] = (__bf16)v;   // V: [b][d][s]
                else
                    OUT[(b * 1024 + p) * D_HEAD + d] = (__bf16)v;   // Q,K: [b][pos][d]
            }
        }
    }
}

// ======================================================================
// Kernel 2: fused flash attention with bias.  One WG = (head b, 128 q rows);
// each wave owns 16 q rows, loops S in BC=64 tiles.  K/V tiles double-
// buffered in LDS via global_load_async_to_lds_b128; bias streamed NT.
// K tile in LDS: [s=64][d=32].  V tile in LDS: [d=32][s=64] (pre-transposed).
// ======================================================================
__global__ __launch_bounds__(256) void attn_kernel(
    const __bf16* __restrict__ qh, const __bf16* __restrict__ kh,
    const __bf16* __restrict__ vh, const float* __restrict__ bias,
    __bf16* __restrict__ oh)
{
    __shared__ __bf16 kbuf[2][BC * D_HEAD];
    __shared__ __bf16 vbuf[2][D_HEAD * BC];
    __shared__ __bf16 pbuf[WAVES][16 * BC];

    const int b     = blockIdx.x;                       // head-batch (n*H+h)
    const int wave  = threadIdx.x >> 5;
    const int lane  = threadIdx.x & 31;
    const int hlf   = lane >> 4;
    const int l16   = lane & 15;
    const int q0    = blockIdx.y * 128 + wave * 16;     // this wave's q rows

    const __bf16* Kb = kh + (size_t)b * S_DIM * D_HEAD;     // [s][d]
    const __bf16* Vb = vh + (size_t)b * D_HEAD * S_DIM;     // [d][s]
    const float*  Bb = bias + (size_t)b * L_DIM * S_DIM;

    // Q A-tile (16x32) lives in registers for the whole S loop
    v16bf qa;
    #pragma unroll
    for (int e = 0; e < 16; ++e)
        qa[e] = qh[((size_t)b * 1024 + q0 + l16) * D_HEAD + a_k(e, hlf)];

    // Async copy of one K/V tile (4 KB each): 256 threads x 16 B
    auto issue_tile = [&](int s0, int bufi) {
        const uint32_t t    = threadIdx.x;
        // K: contiguous 4KB block starting at s0*32 elems
        const uint32_t ldsk  = (uint32_t)(uintptr_t)(&kbuf[bufi][0]) + t * 16;
        const uint32_t goffk = (uint32_t)(s0 * D_HEAD * 2) + t * 16;
        asm volatile("global_load_async_to_lds_b128 %0, %1, %2"
                     :: "v"(ldsk), "v"(goffk), "s"((uint64_t)(uintptr_t)Kb) : "memory");
        // V: 32 rows (d) x 128B, global row stride = S_DIM*2
        const uint32_t row   = t >> 3;                  // d row 0..31
        const uint32_t chunk = t & 7;                   // 16B chunk in row
        const uint32_t ldsv  = (uint32_t)(uintptr_t)(&vbuf[bufi][0]) + t * 16;
        const uint32_t goffv = row * (S_DIM * 2) + (uint32_t)(s0 * 2) + chunk * 16;
        asm volatile("global_load_async_to_lds_b128 %0, %1, %2"
                     :: "v"(ldsv), "v"(goffv), "s"((uint64_t)(uintptr_t)Vb) : "memory");
    };

    v8f oacc[2] = {};                                   // O accum, d tiles 0:16 / 16:32
    float mrow[8], lrow[8];
    #pragma unroll
    for (int r = 0; r < 8; ++r) { mrow[r] = -__builtin_inff(); lrow[r] = 0.f; }

    issue_tile(0, 0);
    const int T = S_DIM / BC;                           // 16 tiles
    for (int it = 0; it < T; ++it) {
        if (it + 1 < T) {
            issue_tile((it + 1) * BC, (it + 1) & 1);
            asm volatile("s_wait_asynccnt 0x2" ::: "memory");   // current tile landed
        } else {
            asm volatile("s_wait_asynccnt 0x0" ::: "memory");
        }
        __syncthreads();
        const __bf16* Kt = &kbuf[it & 1][0];
        const __bf16* Vt = &vbuf[it & 1][0];
        const int s0 = it * BC;

        // ---- scores = Q @ K^T  (+bias, non-temporal) ----
        v8f sc[4];
        #pragma unroll
        for (int nt = 0; nt < 4; ++nt) {
            v16bf bm;                                   // B[k=d][n=s]; contiguous per lane
            #pragma unroll
            for (int e = 0; e < 16; ++e)
                bm[e] = Kt[(nt * 16 + l16) * D_HEAD + e + hlf * 16];
            v8f z = {};
            sc[nt] = wmma_bf16(qa, bm, z);
            #pragma unroll
            for (int r = 0; r < 8; ++r) {
                const int row = q0 + r + hlf * 8;
                sc[nt][r] += __builtin_nontemporal_load(
                    &Bb[(size_t)row * S_DIM + s0 + nt * 16 + l16]);
            }
        }

        // ---- online softmax (rows live across 16 lanes of each half) ----
        #pragma unroll
        for (int r = 0; r < 8; ++r) {
            float mx = fmaxf(fmaxf(sc[0][r], sc[1][r]), fmaxf(sc[2][r], sc[3][r]));
            #pragma unroll
            for (int m = 8; m >= 1; m >>= 1) mx = fmaxf(mx, __shfl_xor(mx, m, 16));
            const float mnew = fmaxf(mrow[r], mx);
            const float corr = __expf(mrow[r] - mnew);
            mrow[r] = mnew;
            float rsum = 0.f;
            #pragma unroll
            for (int nt = 0; nt < 4; ++nt) {
                const float p = __expf(sc[nt][r] - mnew);
                sc[nt][r] = p;
                rsum += p;
            }
            #pragma unroll
            for (int m = 8; m >= 1; m >>= 1) rsum += __shfl_xor(rsum, m, 16);
            lrow[r] = lrow[r] * corr + rsum;
            oacc[0][r] *= corr;
            oacc[1][r] *= corr;
        }

        // ---- C-layout -> A-layout for P via per-wave LDS scratch ----
        __bf16* P = &pbuf[wave][0];
        #pragma unroll
        for (int nt = 0; nt < 4; ++nt)
            #pragma unroll
            for (int r = 0; r < 8; ++r)
                P[(r + hlf * 8) * BC + nt * 16 + l16] = (__bf16)sc[nt][r];

        // ---- O += P @ V ----
        #pragma unroll
        for (int sck = 0; sck < 2; ++sck) {             // s chunks of 32
            v16bf pa;
            #pragma unroll
            for (int e = 0; e < 16; ++e)
                pa[e] = P[l16 * BC + sck * 32 + a_k(e, hlf)];
            #pragma unroll
            for (int nt = 0; nt < 2; ++nt) {            // d tiles of 16
                v16bf bm;                               // B[k=s][n=d]; Vt is [d][s] -> contiguous
                #pragma unroll
                for (int e = 0; e < 16; ++e)
                    bm[e] = Vt[(nt * 16 + l16) * BC + sck * 32 + e + hlf * 16];
                oacc[nt] = wmma_bf16(pa, bm, oacc[nt]);
            }
        }
        __syncthreads();                                // buffer reuse fence
    }

    // ---- finalize: O /= l, store to token-major bf16 [pos*N+n][h*D+d] ----
    const int n = b >> 3, h = b & 7;
    #pragma unroll
    for (int nt = 0; nt < 2; ++nt)
        #pragma unroll
        for (int r = 0; r < 8; ++r) {
            const int p = q0 + r + hlf * 8;
            const float v = oacc[nt][r] / lrow[r];
            oh[((size_t)(p * N_B + n)) * E_DIM + h * D_HEAD + nt * 16 + l16] = (__bf16)v;
        }
}

// ======================================================================
// Kernel 3: out-projection.  d_out[r,j] = O[r,:]·W[j,:] + b[j]   (fp32 out)
// ======================================================================
__global__ __launch_bounds__(256) void out_proj_kernel(
    const __bf16* __restrict__ oh, const float* __restrict__ w,
    const float* __restrict__ bias, float* __restrict__ out)
{
    const int wave = threadIdx.x >> 5;
    const int lane = threadIdx.x & 31;
    const int hlf  = lane >> 4;
    const int l16  = lane & 15;
    const int r0   = blockIdx.x * 128 + wave * 16;

    for (int n0 = 0; n0 < E_DIM; n0 += 64) {
        v8f acc[4] = {};
        for (int k0 = 0; k0 < E_DIM; k0 += 32) {
            v16bf a;
            #pragma unroll
            for (int e = 0; e < 16; ++e)
                a[e] = oh[(size_t)(r0 + l16) * E_DIM + k0 + a_k(e, hlf)];
            #pragma unroll
            for (int nt = 0; nt < 4; ++nt) {
                const int j = n0 + nt * 16 + l16;
                v16bf bm;
                #pragma unroll
                for (int e = 0; e < 16; ++e)
                    bm[e] = (__bf16)w[j * E_DIM + k0 + e + hlf * 16];
                acc[nt] = wmma_bf16(a, bm, acc[nt]);
            }
        }
        #pragma unroll
        for (int nt = 0; nt < 4; ++nt)
            #pragma unroll
            for (int r = 0; r < 8; ++r) {
                const int row = r0 + r + hlf * 8;
                const int col = n0 + nt * 16 + l16;
                out[(size_t)row * E_DIM + col] = acc[nt][r] + bias[col];
            }
    }
}

// ======================================================================
extern "C" void kernel_launch(void* const* d_in, const int* in_sizes, int n_in,
                              void* d_out, int out_size, void* d_ws, size_t ws_size,
                              hipStream_t stream)
{
    const float* query     = (const float*)d_in[0];
    const float* key       = (const float*)d_in[1];
    const float* value     = (const float*)d_in[2];
    const float* attn_bias = (const float*)d_in[3];
    const float* in_w      = (const float*)d_in[4];
    const float* in_b      = (const float*)d_in[5];
    const float* out_w     = (const float*)d_in[6];
    const float* out_b     = (const float*)d_in[7];
    float* out = (float*)d_out;

    const size_t headElems = (size_t)NH * 1024 * D_HEAD;   // 4M elems each
    __bf16* qh = (__bf16*)d_ws;
    __bf16* kh = qh + headElems;
    __bf16* vh = kh + headElems;                           // transposed [b][d][s]
    __bf16* oh = vh + headElems;                           // 16384 x 256

    // 1) QKV projection: 3 x (16384 x 256 x 256) bf16 WMMA GEMM
    qkv_proj_kernel<<<dim3(16384 / 128, 3), 256, 0, stream>>>(
        query, key, value, in_w, in_b, qh, kh, vh);

    // 2) flash attention: 128 heads x 8 q-tiles
    attn_kernel<<<dim3(NH, L_DIM / 128), 256, 0, stream>>>(
        qh, kh, vh, attn_bias, oh);

    // 3) output projection: 16384 x 256 x 256 + bias -> fp32
    out_proj_kernel<<<dim3(16384 / 128), 256, 0, stream>>>(
        oh, out_w, out_b, out);
}
